// MultiHeadAttention_34772055228766
// MI455X (gfx1250) — matrix operand order repaired and checked by reading, not compiled, not run on hardware
//
#include <hip/hip_runtime.h>

// ---------------------------------------------------------------------------
// MHA forward for MI455X (gfx1250), bf16 WMMA path.
//   HIDDEN=1024, B=2, S=2048, 16 heads x 64 dim.
// ---------------------------------------------------------------------------

typedef __attribute__((ext_vector_type(4)))  __bf16 v4bf;
typedef __attribute__((ext_vector_type(8)))  __bf16 v8bf;
typedef __attribute__((ext_vector_type(16))) __bf16 v16bf;
typedef __attribute__((ext_vector_type(8)))  float  v8f;

#define HIDDEN 1024
#define SEQ    2048
#define BATCH  2
#define NHEADS 16
#define HDIM   64
#define MROWS  (BATCH * SEQ)   // 4096

// ---------------------------------------------------------------------------
// f32 -> bf16 conversion (vectorized, optional scale folded in)
// ---------------------------------------------------------------------------
__global__ __launch_bounds__(256) void cvt_f32_bf16(
    const float* __restrict__ src, __bf16* __restrict__ dst, float scale, int n4) {
  int i = blockIdx.x * blockDim.x + threadIdx.x;
  if (i < n4) {
    float4 v = reinterpret_cast<const float4*>(src)[i];
    v4bf o;
    o[0] = (__bf16)(v.x * scale);
    o[1] = (__bf16)(v.y * scale);
    o[2] = (__bf16)(v.z * scale);
    o[3] = (__bf16)(v.w * scale);
    reinterpret_cast<v4bf*>(dst)[i] = o;
  }
}

// ---------------------------------------------------------------------------
// WMMA fragment loaders (contiguous k order k = 16*(lane/16) + s, wave32)
// ---------------------------------------------------------------------------
// A-matrix 16x32 bf16: lane<16 holds row (lane), K = 0..15;
//                      lane>=16 holds row (lane-16), K = 16..31.
__device__ __forceinline__ v16bf load_a_frag(const __bf16* __restrict__ base,
                                             int ld, int row0, int k0, int lane) {
  int r  = row0 + (lane & 15);
  int kk = k0 + ((lane < 16) ? 0 : 16);
  const __bf16* p = base + (size_t)r * ld + kk;
  v8bf lo = *reinterpret_cast<const v8bf*>(p);
  v8bf hi = *reinterpret_cast<const v8bf*>(p + 8);
  v16bf a;
#pragma unroll
  for (int i = 0; i < 8; ++i) { a[i] = lo[i]; a[i + 8] = hi[i]; }
  return a;
}

// B-matrix 32x16 bf16 where B[k][n] = W[n][k] (row-major W, contiguous in k):
// lane<16 -> column n=lane, K=0..15; lane>=16 -> column n=lane-16, K=16..31.
__device__ __forceinline__ v16bf load_b_fragT(const __bf16* __restrict__ w,
                                              int ld, int n0, int k0, int lane) {
  int col = n0 + (lane & 15);
  int kk  = k0 + ((lane < 16) ? 0 : 16);
  return *reinterpret_cast<const v16bf*>(w + (size_t)col * ld + kk);
}

// ---------------------------------------------------------------------------
// Shared GEMM mainloop: acc[4][4] (64x64 wave tile) over K=HIDDEN.
// A [M,1024] bf16 row-major; B = W^T with W [N,1024] bf16 row-major.
// ---------------------------------------------------------------------------
__device__ __forceinline__ void gemm_mainloop(
    const __bf16* __restrict__ A, const __bf16* __restrict__ W,
    int m0, int n0, int lane, v8f (&acc)[4][4]) {
  for (int k0 = 0; k0 < HIDDEN; k0 += 32) {
    v16bf a[4];
#pragma unroll
    for (int i = 0; i < 4; ++i)
      a[i] = load_a_frag(A, HIDDEN, m0 + i * 16, k0, lane);
    if (k0 + 128 < HIDDEN) {   // L0 prefetch for streaming rows (global_prefetch_b8)
      __builtin_prefetch(A + (size_t)(m0 + (lane & 15)) * HIDDEN + k0 + 128, 0, 1);
      __builtin_prefetch(W + (size_t)(n0 + (lane & 15)) * HIDDEN + k0 + 128, 0, 1);
    }
#pragma unroll
    for (int t = 0; t < 4; ++t) {
      v16bf b = load_b_fragT(W, HIDDEN, n0 + t * 16, k0, lane);
#pragma unroll
      for (int i = 0; i < 4; ++i)
        acc[i][t] = __builtin_amdgcn_wmma_f32_16x16x32_bf16(
            false, a[i], false, b, (short)0, acc[i][t], false, false);
    }
  }
}

// ---------------------------------------------------------------------------
// Projection GEMM: Y[4096,1024] = Xbf16 @ Wbf16^T (+ bias*bias_scale)
// Output stored bf16, permuted:
//   vmode==0 : [B, H, S, 64]   (Q, K)
//   vmode==1 : [B, H, 64, S]   (V transposed -> contiguous keys for P@V)
// Block: 128 thr = 4 waves as 2(M) x 2(N); wave tile 64x64; block tile 128x128.
// ---------------------------------------------------------------------------
__global__ __launch_bounds__(128) void gemm_proj(
    const __bf16* __restrict__ A, const __bf16* __restrict__ W,
    const float* __restrict__ bias, float bias_scale,
    __bf16* __restrict__ out, int vmode) {
  const int lane = threadIdx.x & 31;
  const int wid  = threadIdx.x >> 5;
  const int wm = wid & 1, wn = wid >> 1;
  const int m0 = blockIdx.x * 128 + wm * 64;
  const int n0 = blockIdx.y * 128 + wn * 64;

  v8f acc[4][4] = {};
  gemm_mainloop(A, W, m0, n0, lane, acc);

  // C/D layout: VGPR r -> (M = r or r+8 by lane half, N = lane&15)
#pragma unroll
  for (int i = 0; i < 4; ++i) {
#pragma unroll
    for (int t = 0; t < 4; ++t) {
#pragma unroll
      for (int r = 0; r < 8; ++r) {
        int m = m0 + i * 16 + r + ((lane < 16) ? 0 : 8);
        int n = n0 + t * 16 + (lane & 15);
        float v = acc[i][t][r] + bias[n] * bias_scale;
        int b = m >> 11, s = m & (SEQ - 1);
        int h = n >> 6,  d = n & (HDIM - 1);
        size_t idx;
        if (vmode)
          idx = (((size_t)(b * NHEADS + h) * HDIM + d) * SEQ) + s;
        else
          idx = (((size_t)(b * NHEADS + h) * SEQ + s) * HDIM) + d;
        out[idx] = (__bf16)v;
      }
    }
  }
}

// ---------------------------------------------------------------------------
// Flash-attention: one (b,h) per blockIdx.y, 64 q-rows per block (4 waves,
// 16 rows/wave). Streams 64 keys per iteration (16 WMMAs per sync pair);
// online softmax with shfl_xor row reductions; P converted C-layout ->
// A-layout through a per-wave LDS tile. Q pre-scaled by 1/sqrt(64).
// ---------------------------------------------------------------------------
__global__ __launch_bounds__(128) void attn_kernel(
    const __bf16* __restrict__ Q, const __bf16* __restrict__ K,
    const __bf16* __restrict__ VT, __bf16* __restrict__ ctx) {
  __shared__ __bf16 pbuf[4][16 * 64];
  const int lane = threadIdx.x & 31;
  const int wid  = threadIdx.x >> 5;
  const int bh   = blockIdx.y;                 // 0..31
  const int q0   = blockIdx.x * 64 + wid * 16; // q-tile base row

  const __bf16* qbase = Q  + (size_t)bh * SEQ * HDIM;
  const __bf16* kbase = K  + (size_t)bh * SEQ * HDIM;
  const __bf16* vbase = VT + (size_t)bh * HDIM * SEQ;

  v16bf qf[2];
#pragma unroll
  for (int c = 0; c < 2; ++c) qf[c] = load_a_frag(qbase, HDIM, q0, c * 32, lane);

  v8f o[4] = {};
  float mrow[8], lrow[8];
#pragma unroll
  for (int r = 0; r < 8; ++r) { mrow[r] = -1e30f; lrow[r] = 0.f; }

  __bf16* pb = pbuf[wid];

  for (int kb = 0; kb < SEQ; kb += 64) {
    // scores for four 16-key tiles; WMMA K-dim = head dim (2 chunks of 32)
    v8f s[4] = {};
#pragma unroll
    for (int j = 0; j < 4; ++j) {
#pragma unroll
      for (int c = 0; c < 2; ++c) {
        const __bf16* kp = kbase + (size_t)(kb + j * 16 + (lane & 15)) * HDIM
                           + c * 32 + ((lane < 16) ? 0 : 16);
        v16bf kf = *reinterpret_cast<const v16bf*>(kp);
        s[j] = __builtin_amdgcn_wmma_f32_16x16x32_bf16(
            false, qf[c], false, kf, (short)0, s[j], false, false);
      }
    }

    // online softmax over the 16x64 score slab
#pragma unroll
    for (int r = 0; r < 8; ++r) {
      float v = fmaxf(fmaxf(s[0][r], s[1][r]), fmaxf(s[2][r], s[3][r]));
      v = fmaxf(v, __shfl_xor(v, 1));
      v = fmaxf(v, __shfl_xor(v, 2));
      v = fmaxf(v, __shfl_xor(v, 4));
      v = fmaxf(v, __shfl_xor(v, 8));   // row max within each 16-lane half
      float nm    = fmaxf(mrow[r], v);
      float alpha = __expf(mrow[r] - nm);
      mrow[r] = nm;
      float pv[4], rs = 0.f;
#pragma unroll
      for (int j = 0; j < 4; ++j) { pv[j] = __expf(s[j][r] - nm); rs += pv[j]; }
      rs += __shfl_xor(rs, 1);
      rs += __shfl_xor(rs, 2);
      rs += __shfl_xor(rs, 4);
      rs += __shfl_xor(rs, 8);
      lrow[r] = lrow[r] * alpha + rs;
#pragma unroll
      for (int t = 0; t < 4; ++t) o[t][r] *= alpha;
      int row = r + ((lane < 16) ? 0 : 8);
      int col = lane & 15;
#pragma unroll
      for (int j = 0; j < 4; ++j) pb[row * 64 + j * 16 + col] = (__bf16)pv[j];
    }
    __syncthreads();

    // reread P in A-fragment layout (two K=32 chunks), then P @ V
    // (V pre-transposed [B,H,64,S]: keys contiguous per d column)
#pragma unroll
    for (int c = 0; c < 2; ++c) {
      int row = lane & 15;
      int kk  = c * 32 + ((lane < 16) ? 0 : 16);
      v8bf lo = *reinterpret_cast<const v8bf*>(pb + row * 64 + kk);
      v8bf hi = *reinterpret_cast<const v8bf*>(pb + row * 64 + kk + 8);
      v16bf pf;
#pragma unroll
      for (int i = 0; i < 8; ++i) { pf[i] = lo[i]; pf[i + 8] = hi[i]; }
#pragma unroll
      for (int t = 0; t < 4; ++t) {
        const __bf16* vp = vbase + (size_t)(t * 16 + (lane & 15)) * SEQ
                           + kb + c * 32 + ((lane < 16) ? 0 : 16);
        v16bf vf = *reinterpret_cast<const v16bf*>(vp);
        o[t] = __builtin_amdgcn_wmma_f32_16x16x32_bf16(
            false, pf, false, vf, (short)0, o[t], false, false);
      }
    }
    __syncthreads();
  }

  // normalize and store context as bf16 in [B, S, H*64]
  const int b = bh >> 4, h = bh & 15;
#pragma unroll
  for (int r = 0; r < 8; ++r) {
    float inv = 1.0f / lrow[r];
    int srow = q0 + r + ((lane < 16) ? 0 : 8);
#pragma unroll
    for (int t = 0; t < 4; ++t) {
      int d = t * 16 + (lane & 15);
      ctx[((size_t)b * SEQ + srow) * HIDDEN + h * HDIM + d] = (__bf16)(o[t][r] * inv);
    }
  }
}

// ---------------------------------------------------------------------------
// Output GEMM: out[4096,1024] (f32) = ctx_bf16 @ wo_bf16^T + bo
// ---------------------------------------------------------------------------
__global__ __launch_bounds__(128) void gemm_out(
    const __bf16* __restrict__ A, const __bf16* __restrict__ W,
    const float* __restrict__ bias, float* __restrict__ out) {
  const int lane = threadIdx.x & 31;
  const int wid  = threadIdx.x >> 5;
  const int wm = wid & 1, wn = wid >> 1;
  const int m0 = blockIdx.x * 128 + wm * 64;
  const int n0 = blockIdx.y * 128 + wn * 64;

  v8f acc[4][4] = {};
  gemm_mainloop(A, W, m0, n0, lane, acc);

#pragma unroll
  for (int i = 0; i < 4; ++i) {
#pragma unroll
    for (int t = 0; t < 4; ++t) {
#pragma unroll
      for (int r = 0; r < 8; ++r) {
        int m = m0 + i * 16 + r + ((lane < 16) ? 0 : 8);
        int n = n0 + t * 16 + (lane & 15);
        out[(size_t)m * HIDDEN + n] = acc[i][t][r] + bias[n];
      }
    }
  }
}

// ---------------------------------------------------------------------------
// Host launcher
// ---------------------------------------------------------------------------
extern "C" void kernel_launch(void* const* d_in, const int* in_sizes, int n_in,
                              void* d_out, int out_size, void* d_ws, size_t ws_size,
                              hipStream_t stream) {
  const float* x  = (const float*)d_in[0];
  const float* wq = (const float*)d_in[1];
  const float* bq = (const float*)d_in[2];
  const float* wk = (const float*)d_in[3];
  const float* bk = (const float*)d_in[4];
  const float* wv = (const float*)d_in[5];
  const float* bv = (const float*)d_in[6];
  const float* wo = (const float*)d_in[7];
  const float* bo = (const float*)d_in[8];
  float* out = (float*)d_out;

  // workspace layout (bf16 elements)
  __bf16* ws  = (__bf16*)d_ws;
  __bf16* xb  = ws;                                   // 4096*1024
  __bf16* wqb = xb  + (size_t)MROWS * HIDDEN;         // 1024*1024
  __bf16* wkb = wqb + (size_t)HIDDEN * HIDDEN;
  __bf16* wvb = wkb + (size_t)HIDDEN * HIDDEN;
  __bf16* wob = wvb + (size_t)HIDDEN * HIDDEN;
  __bf16* qb  = wob + (size_t)HIDDEN * HIDDEN;        // [B,H,S,64]
  __bf16* kbf = qb  + (size_t)MROWS * HIDDEN;         // [B,H,S,64]
  __bf16* vtb = kbf + (size_t)MROWS * HIDDEN;         // [B,H,64,S]
  __bf16* ctx = vtb + (size_t)MROWS * HIDDEN;         // [B,S,1024]

  const float qscale = 0.125f;  // 1/sqrt(HDIM), folded into wq/bq

  {
    int n4 = MROWS * HIDDEN / 4;
    cvt_f32_bf16<<<(n4 + 255) / 256, 256, 0, stream>>>(x, xb, 1.0f, n4);
    int w4 = HIDDEN * HIDDEN / 4;
    cvt_f32_bf16<<<(w4 + 255) / 256, 256, 0, stream>>>(wq, wqb, qscale, w4);
    cvt_f32_bf16<<<(w4 + 255) / 256, 256, 0, stream>>>(wk, wkb, 1.0f, w4);
    cvt_f32_bf16<<<(w4 + 255) / 256, 256, 0, stream>>>(wv, wvb, 1.0f, w4);
    cvt_f32_bf16<<<(w4 + 255) / 256, 256, 0, stream>>>(wo, wob, 1.0f, w4);
  }

  dim3 gg(MROWS / 128, HIDDEN / 128);   // 32 x 8
  gemm_proj<<<gg, 128, 0, stream>>>(xb, wqb, bq, qscale, qb, 0);
  gemm_proj<<<gg, 128, 0, stream>>>(xb, wkb, bk, 1.0f,  kbf, 0);
  gemm_proj<<<gg, 128, 0, stream>>>(xb, wvb, bv, 1.0f,  vtb, 1);

  attn_kernel<<<dim3(SEQ / 64, BATCH * NHEADS), 128, 0, stream>>>(qb, kbf, vtb, ctx);

  gemm_out<<<gg, 128, 0, stream>>>(ctx, wob, bo, out);
}
